// MixtralAttention_5634997092486
// MI455X (gfx1250) — matrix operand order repaired and checked by reading, not compiled, not run on hardware
//
#include <hip/hip_runtime.h>
#include <hip/hip_bf16.h>

// ---------------------------------------------------------------------------
// MixtralAttention for MI455X (gfx1250, wave32, WMMA).
// Pipeline:
//   1) convert hidden -> f16
//   2) transpose+convert Wqkv, Wo -> f16 (so WMMA B-frags are contiguous)
//   3) qkv = hidden @ Wqkv            (v_wmma_f32_16x16x32_f16,
//        B tile staged in LDS via GLOBAL_LOAD_ASYNC_TO_LDS, double-buffered)
//   4) RoPE(neox) on q,k; split q/k/v to f16 (v stored head-transposed)
//   5) flash attention per (head, 128-query tile), online softmax,
//      QK^T and P@V both on WMMA, P transposed C->A layout via per-wave LDS
//   6) out = attn @ Wo                (same WMMA GEMM)
// ---------------------------------------------------------------------------

typedef __attribute__((ext_vector_type(16))) _Float16 v16h;
typedef __attribute__((ext_vector_type(8)))  _Float16 v8h;
typedef __attribute__((ext_vector_type(8)))  float    v8f;
typedef __attribute__((ext_vector_type(4)))  int      v4i;

#define T_SEQ   2048
#define HID     4096
#define QKV_N   6144
#define NHEADS  32
#define NKV     8
#define HDIM    128
#define SCALE_ATTN 0.08838834764831845f   // 128^-0.5
#define LOG2_THETA 19.931568569324174f    // log2(1e6)

#define WMMA_F16F32(a,b,c) \
  __builtin_amdgcn_wmma_f32_16x16x32_f16(false,(a),false,(b),(short)0,(c),false,false)

// ---- CDNA5 async global->LDS copy (guarded: falls back to ld+ds_store) ----
#if defined(__gfx1250__) && __has_builtin(__builtin_amdgcn_global_load_async_to_lds_b128)
#define HAVE_ASYNC_LDS 1
typedef __attribute__((address_space(1))) v4i* gv4i_p;   // ptr to v4i in global
typedef __attribute__((address_space(3))) v4i* lv4i_p;   // ptr to v4i in LDS
static __device__ __forceinline__ void async_copy16(const void* g, void* l) {
  __builtin_amdgcn_global_load_async_to_lds_b128((gv4i_p)g, (lv4i_p)l, 0, 0);
}
#else
#define HAVE_ASYNC_LDS 0
#endif

static __device__ __forceinline__ void wait_async0() {
#if __has_builtin(__builtin_amdgcn_s_wait_asynccnt)
  __builtin_amdgcn_s_wait_asynccnt(0);
#else
  asm volatile("s_wait_asynccnt 0" ::: "memory");
#endif
}

// ---- WMMA fragment loaders (layouts per CDNA5 ISA 7.12.2, wave32) ---------

// A (16x32 f16), row-major source, stride in elements.
// lane<16: row=lane,   K = {0..7, 16..23}
// lane>=16: row=lane-16, K = {8..15, 24..31}
static __device__ __forceinline__ v16h load_a_frag(const _Float16* __restrict__ base,
                                                   int stride) {
  const int lane = threadIdx.x & 31;
  const int row  = lane & 15;
  const int khi  = lane >> 4;
  const _Float16* p = base + (size_t)row * stride + khi * 8;
  v8h lo = *(const v8h*)(p);
  v8h hi = *(const v8h*)(p + 16);
  return __builtin_shufflevector(lo, hi, 0,1,2,3,4,5,6,7,8,9,10,11,12,13,14,15);
}

// B (32x16 f16) from a column-major tile: element (k,n) at base[n*stride + k].
// lane<16: col=lane, K = {0..7, 16..23} ; lane>=16: col=lane-16, K = {8..15, 24..31}.
static __device__ __forceinline__ v16h load_b_frag(const _Float16* base, int stride) {
  const int lane = threadIdx.x & 31;
  const int col  = lane & 15;
  const int khi  = lane >> 4;
  const _Float16* p = base + (size_t)col * stride + khi * 8;
  v8h lo = *(const v8h*)(p);
  v8h hi = *(const v8h*)(p + 16);
  return __builtin_shufflevector(lo, hi, 0,1,2,3,4,5,6,7,8,9,10,11,12,13,14,15);
}

// ---- elementwise converters ------------------------------------------------

__global__ void convert_f32_to_f16(const float* __restrict__ in,
                                   _Float16* __restrict__ out, size_t n) {
  size_t i = (size_t)blockIdx.x * blockDim.x + threadIdx.x;
  if (i < n) out[i] = (_Float16)in[i];
}

// in: R x C row-major f32 -> out: C x R row-major f16 (i.e. transposed)
__global__ void transpose_to_f16(const float* __restrict__ in,
                                 _Float16* __restrict__ out,
                                 unsigned R, unsigned C) {
  size_t i = (size_t)blockIdx.x * blockDim.x + threadIdx.x;
  size_t total = (size_t)R * C;
  if (i >= total) return;
  unsigned ii = (unsigned)i;
  unsigned r = ii / C, c = ii % C;
  out[(size_t)c * R + r] = (_Float16)in[i];
}

// ---- GEMM: C[M,N](f32) = A[M,K](f16,row) * Bt[N,K](f16,row == B col-major) -
// Block: 256 threads (8 waves), tile 128(M) x 64(N); waves 4(M) x 2(N), each
// wave 32x32 via 4 accumulators. B tile (64x32, 4KB) is staged in LDS with a
// double buffer using async global->LDS DMA; A frags stream from global.
__global__ __launch_bounds__(256) void gemm_f16_wmma(
    const _Float16* __restrict__ A, const _Float16* __restrict__ Bt,
    float* __restrict__ C, int M, int N, int K) {
  __shared__ __align__(32) _Float16 ldsB[2][64 * 32];   // 4KB per buffer

  const int tid  = threadIdx.x;
  const int wave = tid >> 5;
  const int lane = tid & 31;
  const int nb0  = blockIdx.x * 64;                    // block's N origin
  const int m0   = blockIdx.y * 128 + (wave & 3) * 32; // wave rows
  const int wloc = (wave >> 2) * 32;                   // wave's cols in LDS tile
  const int n0   = nb0 + wloc;

  // staging: thread t copies 16B: LDS row sn (=Bt column nb0+sn), chunk sc
  const int sn = tid >> 2;
  const int sc = (tid & 3) * 8;
  const _Float16* bsrc = Bt + (size_t)(nb0 + sn) * K + sc;
  _Float16* sdst0 = &ldsB[0][sn * 32 + sc];
  _Float16* sdst1 = &ldsB[1][sn * 32 + sc];

  v8f acc00 = {}, acc01 = {}, acc10 = {}, acc11 = {};

  const _Float16* a0p = A + (size_t)m0 * K;
  const _Float16* a1p = A + (size_t)(m0 + 16) * K;

  // prologue: stage k=0 into buffer 0
#if HAVE_ASYNC_LDS
  async_copy16(bsrc, sdst0);
  wait_async0();
#else
  *(v8h*)sdst0 = *(const v8h*)bsrc;
#endif
  __syncthreads();

  int buf = 0;
  for (int k = 0; k < K; k += 32) {
    // issue DMA for the next k-slice into the free buffer
    if (k + 32 < K) {
      const _Float16* src = bsrc + k + 32;
      _Float16* dst = buf ? sdst0 : sdst1;
#if HAVE_ASYNC_LDS
      async_copy16(src, dst);
#else
      *(v8h*)dst = *(const v8h*)src;
#endif
    }

    __builtin_prefetch(a0p + k + 256, 0, 1);   // global_prefetch_b8
    v16h a0 = load_a_frag(a0p + k, K);
    v16h a1 = load_a_frag(a1p + k, K);
    v16h b0 = load_b_frag(&ldsB[buf][wloc * 32], 32);
    v16h b1 = load_b_frag(&ldsB[buf][(wloc + 16) * 32], 32);
    acc00 = WMMA_F16F32(a0, b0, acc00);
    acc01 = WMMA_F16F32(a0, b1, acc01);
    acc10 = WMMA_F16F32(a1, b0, acc10);
    acc11 = WMMA_F16F32(a1, b1, acc11);

#if HAVE_ASYNC_LDS
    wait_async0();            // my DMA writes have landed in LDS
#endif
    __syncthreads();          // everyone's writes visible, readers done
    buf ^= 1;
  }

  const int n  = lane & 15;
  const int mb = (lane >> 4) * 8;
#pragma unroll
  for (int r = 0; r < 8; ++r) {
    C[(size_t)(m0      + mb + r) * N + n0      + n] = acc00[r];
    C[(size_t)(m0      + mb + r) * N + n0 + 16 + n] = acc01[r];
    C[(size_t)(m0 + 16 + mb + r) * N + n0      + n] = acc10[r];
    C[(size_t)(m0 + 16 + mb + r) * N + n0 + 16 + n] = acc11[r];
  }
}

// ---- RoPE (neox) + qkv split to f16; v stored transposed per kv-head ------
// grid: (48, T), block: 64.  slot 0..31 = q heads, 32..39 = k heads, 40..47 = v.
__global__ void rope_split_kernel(const float* __restrict__ qkv,
                                  const int* __restrict__ positions,
                                  _Float16* __restrict__ qf,
                                  _Float16* __restrict__ kf,
                                  _Float16* __restrict__ vT) {
  const int i    = threadIdx.x;          // 0..63 (pair i, i+64)
  const int slot = blockIdx.x;           // 0..47
  const int t    = blockIdx.y;           // 0..T-1
  const float inv_freq = exp2f(-(float)i * (LOG2_THETA / 64.0f));
  const float fr = (float)positions[t] * inv_freq;
  float sn, cs;
  __sincosf(fr, &sn, &cs);

  if (slot < 32) {                       // Q
    const float* b = qkv + (size_t)t * QKV_N + slot * HDIM;
    float x1 = b[i], x2 = b[i + 64];
    _Float16* o = qf + (size_t)t * HID + slot * HDIM;
    o[i]      = (_Float16)(x1 * cs - x2 * sn);
    o[i + 64] = (_Float16)(x2 * cs + x1 * sn);
  } else if (slot < 40) {                // K
    const int kvh = slot - 32;
    const float* b = qkv + (size_t)t * QKV_N + HID + kvh * HDIM;
    float x1 = b[i], x2 = b[i + 64];
    _Float16* o = kf + (size_t)t * (NKV * HDIM) + kvh * HDIM;
    o[i]      = (_Float16)(x1 * cs - x2 * sn);
    o[i + 64] = (_Float16)(x2 * cs + x1 * sn);
  } else {                               // V (no rope), transpose to [kvh*128+d][t]
    const int kvh = slot - 40;
    const float* b = qkv + (size_t)t * QKV_N + HID + NKV * HDIM + kvh * HDIM;
    vT[(size_t)(kvh * HDIM + i)      * T_SEQ + t] = (_Float16)b[i];
    vT[(size_t)(kvh * HDIM + i + 64) * T_SEQ + t] = (_Float16)b[i + 64];
  }
}

// ---- Flash attention: grid (T/128, NHEADS), block 256 (8 waves) -----------
// Each wave owns 16 query rows; causal online softmax over 32-key blocks.
// No block barriers (trip counts diverge per wave); per-wave LDS only.
__global__ __launch_bounds__(256) void flash_attn_wmma(
    const _Float16* __restrict__ qf, const _Float16* __restrict__ kf,
    const _Float16* __restrict__ vT, _Float16* __restrict__ attn_out) {
  __shared__ __align__(32) _Float16 ldsP[8 * 512];   // 1KB per wave: 16x32 P tile

  const int wave = threadIdx.x >> 5;
  const int lane = threadIdx.x & 31;
  const int head = blockIdx.y;
  const int kvh  = head >> 2;            // 32 q heads / 8 kv heads
  const int qw   = blockIdx.x * 128 + wave * 16;   // first query row of wave
  const int n    = lane & 15;
  const int h    = lane >> 4;

  // Q fragments: 16 rows x 128 d, as 4 A-frags (K chunks of 32)
  v16h qa[4];
#pragma unroll
  for (int kc = 0; kc < 4; ++kc)
    qa[kc] = load_a_frag(qf + (size_t)qw * HID + head * HDIM + kc * 32, HID);

  v8f O[8];
#pragma unroll
  for (int dc = 0; dc < 8; ++dc) O[dc] = (v8f){};
  float mrow[8], lrow[8];
#pragma unroll
  for (int r = 0; r < 8; ++r) { mrow[r] = -1e30f; lrow[r] = 0.0f; }

  _Float16* myP = &ldsP[wave * 512];

  for (int s0 = 0; s0 < qw + 16; s0 += 32) {       // uniform per wave
    // ---- S = Q K^T for keys [s0, s0+32) : 8 WMMAs ----
    v8f S0 = {}, S1 = {};
#pragma unroll
    for (int kc = 0; kc < 4; ++kc) {
      v16h b0 = load_b_frag(kf + (size_t)s0 * (NKV * HDIM) + kvh * HDIM + kc * 32,
                            NKV * HDIM);
      v16h b1 = load_b_frag(kf + (size_t)(s0 + 16) * (NKV * HDIM) + kvh * HDIM + kc * 32,
                            NKV * HDIM);
      S0 = WMMA_F16F32(qa[kc], b0, S0);
      S1 = WMMA_F16F32(qa[kc], b1, S1);
    }

    // ---- scale + causal mask + block row-max ----
    float a0[8], a1[8], bm[8];
#pragma unroll
    for (int r = 0; r < 8; ++r) {
      const int query = qw + r + 8 * h;
      float s0v = S0[r] * SCALE_ATTN;
      float s1v = S1[r] * SCALE_ATTN;
      if (s0 + n      > query) s0v = -1e30f;
      if (s0 + 16 + n > query) s1v = -1e30f;
      a0[r] = s0v; a1[r] = s1v;
      bm[r] = fmaxf(s0v, s1v);
    }
#pragma unroll
    for (int off = 1; off < 16; off <<= 1)
#pragma unroll
      for (int r = 0; r < 8; ++r)
        bm[r] = fmaxf(bm[r], __shfl_xor(bm[r], off, 32));

    // ---- online softmax update; write P to per-wave LDS (C->A relayout) ----
    float alph[8], rsum[8];
#pragma unroll
    for (int r = 0; r < 8; ++r) {
      const float mnew = fmaxf(mrow[r], bm[r]);
      alph[r] = __expf(mrow[r] - mnew);
      const float p0 = __expf(a0[r] - mnew);
      const float p1 = __expf(a1[r] - mnew);
      rsum[r] = p0 + p1;
      mrow[r] = mnew;
      myP[(r + 8 * h) * 32 + n]      = (_Float16)p0;
      myP[(r + 8 * h) * 32 + 16 + n] = (_Float16)p1;
    }
#pragma unroll
    for (int off = 1; off < 16; off <<= 1)
#pragma unroll
      for (int r = 0; r < 8; ++r)
        rsum[r] += __shfl_xor(rsum[r], off, 32);
#pragma unroll
    for (int r = 0; r < 8; ++r) lrow[r] = lrow[r] * alph[r] + rsum[r];

    // rescale running output
#pragma unroll
    for (int dc = 0; dc < 8; ++dc)
#pragma unroll
      for (int r = 0; r < 8; ++r) O[dc][r] *= alph[r];

    // ---- O += P @ V  (A-frag from LDS, B-frags from transposed V) ----
    {
      const int row = lane & 15;
      const int khi = lane >> 4;
      const _Float16* pl = myP + row * 32 + khi * 8;
      v8h lo = *(const v8h*)(pl);
      v8h hi = *(const v8h*)(pl + 16);
      v16h pA = __builtin_shufflevector(lo, hi, 0,1,2,3,4,5,6,7,8,9,10,11,12,13,14,15);
#pragma unroll
      for (int dc = 0; dc < 8; ++dc) {
        v16h vB = load_b_frag(vT + (size_t)(kvh * HDIM + dc * 16) * T_SEQ + s0, T_SEQ);
        O[dc] = WMMA_F16F32(pA, vB, O[dc]);
      }
    }
  }

  // ---- epilogue: O /= l, store f16 ----
  float invl[8];
#pragma unroll
  for (int r = 0; r < 8; ++r) invl[r] = 1.0f / lrow[r];
#pragma unroll
  for (int dc = 0; dc < 8; ++dc)
#pragma unroll
    for (int r = 0; r < 8; ++r)
      attn_out[(size_t)(qw + r + 8 * h) * HID + head * HDIM + dc * 16 + n] =
          (_Float16)(O[dc][r] * invl[r]);
}

// ---------------------------------------------------------------------------
// host launcher
// ---------------------------------------------------------------------------
extern "C" void kernel_launch(void* const* d_in, const int* in_sizes, int n_in,
                              void* d_out, int out_size, void* d_ws, size_t ws_size,
                              hipStream_t stream) {
  const int*   positions = (const int*)d_in[0];
  const float* hidden    = (const float*)d_in[1];
  const float* Wqkv      = (const float*)d_in[2];
  const float* Wo        = (const float*)d_in[3];
  float*       out       = (float*)d_out;

  char* ws = (char*)d_ws;
  _Float16* hF    = (_Float16*)(ws + 0);            // 2048x4096 f16   (16 MB)
  _Float16* WqkvT = (_Float16*)(ws + 16777216);     // 6144x4096 f16   (48 MB)
  _Float16* WoT   = (_Float16*)(ws + 67108864);     // 4096x4096 f16   (32 MB)
  float*    qkv   = (float*)   (ws + 100663296);    // 2048x6144 f32   (48 MB)
  _Float16* qF    = (_Float16*)(ws + 150994944);    // 2048x4096 f16   (16 MB)
  _Float16* kF    = (_Float16*)(ws + 167772160);    // 2048x1024 f16   ( 4 MB)
  _Float16* vT    = (_Float16*)(ws + 171966464);    // 1024x2048 f16   ( 4 MB)
  _Float16* attnF = (_Float16*)(ws + 176160768);    // 2048x4096 f16   (16 MB)

  // 1) hidden -> f16
  {
    size_t n = (size_t)T_SEQ * HID;
    convert_f32_to_f16<<<dim3((unsigned)((n + 255) / 256)), dim3(256), 0, stream>>>(hidden, hF, n);
  }
  // 2) transpose+convert weights
  {
    size_t n = (size_t)HID * QKV_N;
    transpose_to_f16<<<dim3((unsigned)((n + 255) / 256)), dim3(256), 0, stream>>>(Wqkv, WqkvT, HID, QKV_N);
  }
  {
    size_t n = (size_t)HID * HID;
    transpose_to_f16<<<dim3((unsigned)((n + 255) / 256)), dim3(256), 0, stream>>>(Wo, WoT, HID, HID);
  }
  // 3) qkv = hidden @ Wqkv
  gemm_f16_wmma<<<dim3(QKV_N / 64, T_SEQ / 128), dim3(256), 0, stream>>>(
      hF, WqkvT, qkv, T_SEQ, QKV_N, HID);
  // 4) RoPE + split
  rope_split_kernel<<<dim3(48, T_SEQ), dim3(64), 0, stream>>>(qkv, positions, qF, kF, vT);
  // 5) flash attention
  flash_attn_wmma<<<dim3(T_SEQ / 128, NHEADS), dim3(256), 0, stream>>>(qF, kF, vT, attnF);
  // 6) out = attn @ Wo
  gemm_f16_wmma<<<dim3(HID / 64, T_SEQ / 128), dim3(256), 0, stream>>>(
      attnF, WoT, out, T_SEQ, HID, HID);
}
